// DAGNN_14594298872388
// MI455X (gfx1250) — compile-verified, matrix operand order checked by reading, not run on hardware
//
#include <hip/hip_runtime.h>
#include <hip/hip_bf16.h>

#define IN_DIM 500
#define HID    256
#define NCLASS 50
#define KHOPS  10

typedef __attribute__((ext_vector_type(16))) _Float16 v16h;
typedef __attribute__((ext_vector_type(8)))  _Float16 h8;
typedef __attribute__((ext_vector_type(4)))  _Float16 h4;
typedef __attribute__((ext_vector_type(8)))  float    v8f;

// Build a 16-bit WMMA A/B fragment from an LDS tile row (ISA 7.12.2 layout):
// lane%16 selects M (or N), lane/16 selects the 8-wide K sub-group; v16h
// elements 0..7 hold K = kb..kb+7, elements 8..15 hold K = 16+kb..16+kb+7.
__device__ __forceinline__ v16h load_frag(const _Float16* row, int kb) {
  h8 lo = *(const h8*)(row + kb);
  h8 hi = *(const h8*)(row + 16 + kb);
  v16h f;
#pragma unroll
  for (int i = 0; i < 8; ++i) { f[i] = lo[i]; f[8 + i] = hi[i]; }
  return f;
}

__device__ __forceinline__ v8f wmma16(v16h a, v16h b, v8f c) {
  return __builtin_amdgcn_wmma_f32_16x16x32_f16(false, a, false, b, (short)0,
                                                c, false, false);
}

// ---------------- GEMM1: relu(feats @ W1 + b1) -> hid (f16) -----------------
// Block tile 64x64, 8 waves (4M x 2N), each wave 16x32 (two WMMAs sharing A).
// Double-buffered LDS; branchless clamped b128 global loads.
__global__ void __launch_bounds__(256)
dagnn_gemm1(const float* __restrict__ X, const float* __restrict__ W,
            const float* __restrict__ bias, _Float16* __restrict__ Y, int Nn) {
  // stride 40 halves: rows 16B-aligned and bank-conflict-free for frag reads
  __shared__ __align__(16) _Float16 As[2][64][40];
  __shared__ __align__(16) _Float16 Bt[2][64][40];   // W1 tile, transposed [n][k]
  const int t = threadIdx.x;
  const int wave = t >> 5, lane = t & 31;
  const int waveM = wave >> 1, waveN = wave & 1;
  const int blockM = blockIdx.x * 64;
  const int blockN = blockIdx.y * 64;

  // ---- tile loaders (branchless: clamp address, cndmask value) ----
  auto loadA = [&](int kt, int buf) {
#pragma unroll
    for (int j = 0; j < 2; ++j) {
      int idx = j * 256 + t;               // 512 float4 groups (64 x 8)
      int r = idx >> 3, k4 = (idx & 7) * 4;
      int grow = blockM + r, gk = kt + k4;
      int cr = grow < Nn ? grow : Nn - 1;          // clamp row (garbage rows never stored)
      int ck = gk < IN_DIM ? gk : IN_DIM - 4;      // clamp k
      float4 v = *(const float4*)(X + (size_t)cr * IN_DIM + ck);
      bool ok = gk < IN_DIM;                       // K-tail must be exactly zero
      h4 hv;
      hv[0] = ok ? (_Float16)v.x : (_Float16)0.f;
      hv[1] = ok ? (_Float16)v.y : (_Float16)0.f;
      hv[2] = ok ? (_Float16)v.z : (_Float16)0.f;
      hv[3] = ok ? (_Float16)v.w : (_Float16)0.f;
      *(h4*)&As[buf][r][k4] = hv;
    }
  };
  auto loadB = [&](int kt, int buf) {
#pragma unroll
    for (int j = 0; j < 2; ++j) {
      int idx = j * 256 + t;               // 512 float4 groups (32k x 16 n-groups)
      int kk = idx >> 4, n4 = (idx & 15) * 4;
      int gk = kt + kk;
      int ck = gk < IN_DIM ? gk : IN_DIM - 1;
      float4 v = *(const float4*)(W + (size_t)ck * HID + blockN + n4);
      bool ok = gk < IN_DIM;
      Bt[buf][n4 + 0][kk] = ok ? (_Float16)v.x : (_Float16)0.f;
      Bt[buf][n4 + 1][kk] = ok ? (_Float16)v.y : (_Float16)0.f;
      Bt[buf][n4 + 2][kk] = ok ? (_Float16)v.z : (_Float16)0.f;
      Bt[buf][n4 + 3][kk] = ok ? (_Float16)v.w : (_Float16)0.f;
    }
  };

  v8f acc0 = {0.f, 0.f, 0.f, 0.f, 0.f, 0.f, 0.f, 0.f};
  v8f acc1 = acc0;
  const int mrow = waveM * 16 + (lane & 15);
  const int nrow = waveN * 32 + (lane & 15);
  const int kb = (lane >> 4) * 8;

  loadA(0, 0);
  loadB(0, 0);
  __syncthreads();
  const int NKT = 16;                       // 500 padded to 512
  for (int it = 0; it < NKT; ++it) {
    int cur = it & 1;
    if (it + 1 < NKT) {                     // prefetch next tile into other buffer
      loadA((it + 1) * 32, cur ^ 1);
      loadB((it + 1) * 32, cur ^ 1);
    }
    v16h a  = load_frag(&As[cur][mrow][0], kb);
    v16h b0 = load_frag(&Bt[cur][nrow][0], kb);
    v16h b1 = load_frag(&Bt[cur][nrow + 16][0], kb);
    acc0 = wmma16(a, b0, acc0);
    acc1 = wmma16(a, b1, acc1);
    __syncthreads();
  }

  const int col0 = blockN + waveN * 32 + (lane & 15);
  const int col1 = col0 + 16;
  const int rbase = blockM + waveM * 16 + (lane >> 4) * 8;
  const float bv0 = bias[col0], bv1 = bias[col1];
#pragma unroll
  for (int i = 0; i < 8; ++i) {            // C layout: VGPR i -> M = i + 8*(lane/16)
    int row = rbase + i;
    if (row < Nn) {
      float v0 = acc0[i] + bv0;
      float v1 = acc1[i] + bv1;
      Y[(size_t)row * HID + col0] = (_Float16)(v0 > 0.f ? v0 : 0.f);
      Y[(size_t)row * HID + col1] = (_Float16)(v1 > 0.f ? v1 : 0.f);
    }
  }
}

// ---------------- GEMM2: hid @ W2 + b2 -> plane0 (f32) ----------------------
// Block tile 32x64 (N padded 50->64), 8 waves (2M x 4N), double-buffered.
__global__ void __launch_bounds__(256)
dagnn_gemm2(const _Float16* __restrict__ Hd, const float* __restrict__ W2,
            const float* __restrict__ b2, float* __restrict__ plane0, int Nn) {
  __shared__ __align__(16) _Float16 As[2][32][40];
  __shared__ __align__(16) _Float16 Bt[2][64][40];
  const int t = threadIdx.x;
  const int wave = t >> 5, lane = t & 31;
  const int waveM = wave >> 2, waveN = wave & 3;
  const int blockM = blockIdx.x * 32;

  auto loadA = [&](int kt, int buf) {      // 32x32 halves, already f16: h4 copies
    int r = t >> 3, k4 = (t & 7) * 4;
    int grow = blockM + r;
    int cr = grow < Nn ? grow : Nn - 1;    // clamped rows never stored; K always valid
    *(h4*)&As[buf][r][k4] = *(const h4*)(Hd + (size_t)cr * HID + kt + k4);
  };
  auto loadB = [&](int kt, int buf) {      // 32k x 64n, cols >= 50 zero
#pragma unroll
    for (int j = 0; j < 8; ++j) {
      int idx = j * 256 + t;
      int n = idx & 63, kk = idx >> 6;
      int cn = n < NCLASS ? n : NCLASS - 1;
      float v = W2[(size_t)(kt + kk) * NCLASS + cn];
      Bt[buf][n][kk] = (n < NCLASS) ? (_Float16)v : (_Float16)0.f;
    }
  };

  v8f acc = {0.f, 0.f, 0.f, 0.f, 0.f, 0.f, 0.f, 0.f};
  const int mrow = waveM * 16 + (lane & 15);
  const int nrow = waveN * 16 + (lane & 15);
  const int kb = (lane >> 4) * 8;

  loadA(0, 0);
  loadB(0, 0);
  __syncthreads();
  const int NKT = HID / 32;                // 8, K = 256 exactly
  for (int it = 0; it < NKT; ++it) {
    int cur = it & 1;
    if (it + 1 < NKT) {
      loadA((it + 1) * 32, cur ^ 1);
      loadB((it + 1) * 32, cur ^ 1);
    }
    v16h a = load_frag(&As[cur][mrow][0], kb);
    v16h b = load_frag(&Bt[cur][nrow][0], kb);
    acc = wmma16(a, b, acc);
    __syncthreads();
  }

  const int col = waveN * 16 + (lane & 15);
  const int rbase = blockM + waveM * 16 + (lane >> 4) * 8;
  if (col < NCLASS) {
    const float bv = b2[col];
#pragma unroll
    for (int i = 0; i < 8; ++i) {
      int row = rbase + i;
      if (row < Nn) plane0[(size_t)row * NCLASS + col] = acc[i] + bv;
    }
  }
}

// ---------------- Graph propagation ----------------------------------------
__global__ void dagnn_degree(const int* __restrict__ dst, float* __restrict__ deg, int E) {
  int i = blockIdx.x * blockDim.x + threadIdx.x;
  if (i < E) atomicAdd(&deg[dst[i]], 1.0f);
}

__global__ void dagnn_norm(float* __restrict__ deg, int Nn) {
  int i = blockIdx.x * blockDim.x + threadIdx.x;
  if (i < Nn) deg[i] = rsqrtf(deg[i]);   // self-loops guarantee deg >= 1
}

// One wave per edge: lane = channel (c and c+32); out[dst] += norm[src]*in[src].
__global__ void dagnn_scatter(const float* __restrict__ in, float* __restrict__ outp,
                              const float* __restrict__ norm,
                              const int* __restrict__ src, const int* __restrict__ dst,
                              int E) {
  int gid = blockIdx.x * blockDim.x + threadIdx.x;
  int e = gid >> 5, lane = gid & 31;
  if (e >= E) return;
  int sv = src[e], dv = dst[e];
  float ns = norm[sv];
  const float* ip = in + (size_t)sv * NCLASS;
  float* op = outp + (size_t)dv * NCLASS;
  atomicAdd(&op[lane], ns * ip[lane]);
  if (lane < NCLASS - 32)
    atomicAdd(&op[lane + 32], ns * ip[lane + 32]);
}

__global__ void dagnn_scale(float* __restrict__ p, const float* __restrict__ norm, int Nn) {
  int i = blockIdx.x * blockDim.x + threadIdx.x;
  if (i < Nn * NCLASS) p[i] *= norm[i / NCLASS];
}

// ---------------- Hop attention: S = sigmoid(H s); out = sum_k S_k H_k ------
__global__ void dagnn_combine(const float* __restrict__ planes, const float* __restrict__ s,
                              float* __restrict__ out, int Nn) {
  int gid = blockIdx.x * blockDim.x + threadIdx.x;
  int node = gid >> 5, lane = gid & 31;
  if (node >= Nn) return;
  int c2 = lane + 32;
  float s1 = s[lane];
  float s2 = (c2 < NCLASS) ? s[c2] : 0.f;
  float acc1 = 0.f, acc2 = 0.f;
  size_t pstride = (size_t)Nn * NCLASS;
  size_t base = (size_t)node * NCLASS;
#pragma unroll
  for (int k = 0; k <= KHOPS; ++k) {
    const float* hp = planes + (size_t)k * pstride + base;
    float h1 = hp[lane];
    float h2 = (c2 < NCLASS) ? hp[c2] : 0.f;
    float z = h1 * s1 + h2 * s2;
#pragma unroll
    for (int off = 16; off > 0; off >>= 1) z += __shfl_xor(z, off, 32);
    float S = 1.0f / (1.0f + __expf(-z));
    acc1 += S * h1;
    acc2 += S * h2;
  }
  out[base + lane] = acc1;
  if (c2 < NCLASS) out[base + c2] = acc2;
}

extern "C" void kernel_launch(void* const* d_in, const int* in_sizes, int n_in,
                              void* d_out, int out_size, void* d_ws, size_t ws_size,
                              hipStream_t stream) {
  const float* feats = (const float*)d_in[0];
  const float* W1    = (const float*)d_in[1];
  const float* b1    = (const float*)d_in[2];
  const float* W2    = (const float*)d_in[3];
  const float* b2    = (const float*)d_in[4];
  const float* s     = (const float*)d_in[5];
  const int*   src   = (const int*)d_in[6];
  const int*   dst   = (const int*)d_in[7];
  const int Nn = in_sizes[0] / IN_DIM;   // 100000
  const int E  = in_sizes[6];            // 1.6M
  float* out = (float*)d_out;

  // Workspace carve: norm (N f32), hid (N x 256 f16), planes ((K+1) x N x 50 f32)
  char* ws = (char*)d_ws;
  size_t off = 0;
  auto carve = [&](size_t bytes) -> char* {
    char* p = ws + off;
    off = (off + bytes + 255) & ~(size_t)255;
    return p;
  };
  float*    norm   = (float*)   carve((size_t)Nn * sizeof(float));
  _Float16* hid    = (_Float16*)carve((size_t)Nn * HID * sizeof(_Float16));
  float*    planes = (float*)   carve((size_t)(KHOPS + 1) * Nn * NCLASS * sizeof(float));
  if (off > ws_size) return;  // insufficient scratch

  const size_t pstride = (size_t)Nn * NCLASS;

  // degrees -> deg^{-1/2}
  hipMemsetAsync(norm, 0, (size_t)Nn * sizeof(float), stream);
  dagnn_degree<<<(E + 255) / 256, 256, 0, stream>>>(dst, norm, E);
  dagnn_norm<<<(Nn + 255) / 256, 256, 0, stream>>>(norm, Nn);

  // MLP via WMMA
  dim3 g1((Nn + 63) / 64, HID / 64);
  dagnn_gemm1<<<g1, 256, 0, stream>>>(feats, W1, b1, hid, Nn);
  dim3 g2((Nn + 31) / 32, 1);
  dagnn_gemm2<<<g2, 256, 0, stream>>>(hid, W2, b2, planes, Nn);

  // K-hop symmetric-normalized propagation
  hipMemsetAsync(planes + pstride, 0, (size_t)KHOPS * pstride * sizeof(float), stream);
  const int sgrid = (int)(((size_t)E * 32 + 255) / 256);
  const int cgrid = (Nn * NCLASS + 255) / 256;
  for (int k = 1; k <= KHOPS; ++k) {
    const float* pin = planes + (size_t)(k - 1) * pstride;
    float* pout      = planes + (size_t)k * pstride;
    dagnn_scatter<<<sgrid, 256, 0, stream>>>(pin, pout, norm, src, dst, E);
    dagnn_scale<<<cgrid, 256, 0, stream>>>(pout, norm, Nn);
  }

  // hop attention + readout
  dagnn_combine<<<(int)(((size_t)Nn * 32 + 255) / 256), 256, 0, stream>>>(planes, s, out, Nn);
  (void)n_in; (void)out_size;
}
